// MultiHeadAttention_30562987278547
// MI455X (gfx1250) — matrix operand order repaired and checked by reading, not compiled, not run on hardware
//
#include <hip/hip_runtime.h>

#define BB 2
#define SS 2048
#define DD 1024
#define HH 16
#define HD 64
#define MM (BB*SS)   // 4096

typedef __bf16 bf16;
typedef __attribute__((ext_vector_type(16))) __bf16 v16bf;
typedef __attribute__((ext_vector_type(8)))  float  v8f;
typedef __attribute__((ext_vector_type(4)))  float  v4f;
typedef __attribute__((ext_vector_type(4)))  unsigned int v4u;

union ABFrag { v4u u[2]; v16bf v; };
union Pack8  { __bf16 h[8]; v4u u; };
union Pack2  { __bf16 h[2]; unsigned int u; };

// ---- fragment loaders (WMMA 16x16x32 bf16, contiguous k = 16g+s used by both A and B) ----

// A matrix 16x32: lane holds row (lane&15); lane-group g holds K = k0+16g .. +15 (one 32B run)
__device__ __forceinline__ v16bf load_a_bf16(const bf16* base, int ld, int row, int k0, int lane) {
  int m = row + (lane & 15);
  int g = (lane >> 4) & 1;
  const bf16* p = base + (size_t)m * ld + k0 + g * 16;
  ABFrag f;
  f.u[0] = *(const v4u*)p;
  f.u[1] = *(const v4u*)(p + 8);
  return f.v;
}

// B matrix 32x16 from column-major storage (Bt row-major [N][K]):
// lane holds col (lane&15); lane-group g holds K = k0+16g .. +15 (one 32B run)
__device__ __forceinline__ v16bf load_b_bf16(const bf16* baseT, int ld, int col, int k0, int lane) {
  int n = col + (lane & 15);
  int g = (lane >> 4) & 1;
  const bf16* p = baseT + (size_t)n * ld + k0 + g * 16;
  ABFrag f;
  f.u[0] = *(const v4u*)p;
  f.u[1] = *(const v4u*)(p + 8);
  return f.v;
}

__device__ __forceinline__ v8f wmma_bf16(v16bf a, v16bf b, v8f c) {
  return __builtin_amdgcn_wmma_f32_16x16x32_bf16(false, a, false, b, (short)0, c, false, false);
}

// CDNA5 async memory->LDS copy (ISA 15.18.3, tracked with ASYNCcnt).
// lds_off = LDS byte address (low 32 bits of the flat shared address).
__device__ __forceinline__ void async_load_b128(unsigned lds_off, const bf16* gptr) {
  asm volatile("global_load_async_to_lds_b128 %0, %1, off"
               :: "v"(lds_off), "v"(gptr)
               : "memory");
}
__device__ __forceinline__ unsigned lds_addr(const void* p) {
  return (unsigned)(unsigned long long)p;
}

// ---- weight transpose + bf16 convert: Wt[n][k] = (bf16)W[k][n] ----
__global__ __launch_bounds__(256) void wconv_kernel(const float* __restrict__ W,
                                                    bf16* __restrict__ Wt) {
  __shared__ float tile[32][33];
  int k0 = blockIdx.x * 32, n0 = blockIdx.y * 32;
  int tx = threadIdx.x & 31, ty = threadIdx.x >> 5;
#pragma unroll
  for (int i = 0; i < 32; i += 8)
    tile[ty + i][tx] = W[(size_t)(k0 + ty + i) * DD + n0 + tx];
  __syncthreads();
#pragma unroll
  for (int i = 0; i < 32; i += 8)
    Wt[(size_t)(n0 + ty + i) * DD + k0 + tx] = (bf16)tile[tx][ty + i];
}

// ---- GEMM: Out = X @ W + bias ----
// Block = 8 waves (2x4), block tile 128x128, wave tile 64x32, k-chunk 32.
// A (128x32, converted to bf16 once) and B (128x32) staged per chunk into
// double-buffered LDS; register-pipelined global loads overlap the 8 WMMAs.
// IN_BF16: 0 = fp32 X (convert while staging), 1 = bf16 X
// OUT_MODE: 0 = bf16 [M][N]; 1 = bf16 head-transposed [B][H][HD][S]; 2 = fp32 [M][N]
template<int IN_BF16, int OUT_MODE>
__global__ __launch_bounds__(256) void gemm_kernel(const void* __restrict__ Xv,
    const bf16* __restrict__ Wt, const float* __restrict__ bias,
    void* __restrict__ Out, int Mdim, int Ndim, int Kdim) {
  __shared__ __align__(16) bf16 alds[2][128][32];  // 16 KB
  __shared__ __align__(16) bf16 blds[2][128][32];  // 16 KB (Bt layout: [n][k])
  int tid = threadIdx.x;
  int lane = tid & 31;
  int w = tid >> 5;
  int waveM = w >> 2, waveN = w & 3;  // 2 x 4 wave grid
  int gRow = blockIdx.x * 128 + waveM * 64;
  int gCol = blockIdx.y * 128 + waveN * 32;
  int rowBase = waveM * 64;   // LDS-relative
  int colBase = waveN * 32;
  int n15 = lane & 15, g = (lane >> 4) & 1;

  // staging coords: each thread owns one (row, 16-wide k-half) = 32 B of each tile
  int srow = tid >> 1, shalf = (tid & 1) * 16;

  v8f zero = {};
  v8f acc[4][2];
#pragma unroll
  for (int i = 0; i < 4; ++i)
#pragma unroll
    for (int j = 0; j < 2; ++j) acc[i][j] = zero;

  // register staging buffers
  v4f af[4];  // fp32 path: 16 floats
  v4u au[2];  // bf16 path: 16 bf16
  v4u bu[2];

  auto stage_load = [&](int k0) {
    if (IN_BF16) {
      const bf16* p = (const bf16*)Xv + (size_t)(blockIdx.x * 128 + srow) * Kdim + k0 + shalf;
      au[0] = *(const v4u*)p;
      au[1] = *(const v4u*)(p + 8);
    } else {
      const float* p = (const float*)Xv + (size_t)(blockIdx.x * 128 + srow) * Kdim + k0 + shalf;
      af[0] = *(const v4f*)p;
      af[1] = *(const v4f*)(p + 4);
      af[2] = *(const v4f*)(p + 8);
      af[3] = *(const v4f*)(p + 12);
    }
    const bf16* q = Wt + (size_t)(blockIdx.y * 128 + srow) * Kdim + k0 + shalf;
    bu[0] = *(const v4u*)q;
    bu[1] = *(const v4u*)(q + 8);
  };
  auto stage_store = [&](int buf) {
    bf16* ad = &alds[buf][srow][shalf];
    if (IN_BF16) {
      *(v4u*)ad = au[0];
      *(v4u*)(ad + 8) = au[1];
    } else {
      Pack8 p0, p1;
#pragma unroll
      for (int i = 0; i < 4; ++i) {
        p0.h[i]     = (bf16)af[0][i];
        p0.h[4 + i] = (bf16)af[1][i];
        p1.h[i]     = (bf16)af[2][i];
        p1.h[4 + i] = (bf16)af[3][i];
      }
      *(v4u*)ad = p0.u;
      *(v4u*)(ad + 8) = p1.u;
    }
    bf16* bd = &blds[buf][srow][shalf];
    *(v4u*)bd = bu[0];
    *(v4u*)(bd + 8) = bu[1];
  };

  stage_load(0);
  const int nch = Kdim / 32;
  for (int i = 0; i < nch; ++i) {
    int buf = i & 1;
    stage_store(buf);
    __syncthreads();
    if (i + 1 < nch) stage_load((i + 1) * 32);  // overlaps the WMMA work below

    v16bf a[4], bfr[2];
#pragma unroll
    for (int fi = 0; fi < 4; ++fi)
      a[fi] = load_a_bf16(&alds[buf][0][0], 32, rowBase + fi * 16, 0, lane);
#pragma unroll
    for (int fj = 0; fj < 2; ++fj)
      bfr[fj] = load_b_bf16(&blds[buf][0][0], 32, colBase + fj * 16, 0, lane);
#pragma unroll
    for (int fi = 0; fi < 4; ++fi)
#pragma unroll
      for (int fj = 0; fj < 2; ++fj)
        acc[fi][fj] = wmma_bf16(a[fi], bfr[fj], acc[fi][fj]);
  }

#pragma unroll
  for (int i = 0; i < 4; ++i) {
#pragma unroll
    for (int j = 0; j < 2; ++j) {
      int nf = gCol + j * 16 + n15;
      float bv = bias[nf];
      if (OUT_MODE == 0) {
        bf16* dst = (bf16*)Out + (size_t)(gRow + i * 16 + 8 * g) * Ndim + nf;
#pragma unroll
        for (int r = 0; r < 8; ++r)
          dst[(size_t)r * Ndim] = (bf16)(acc[i][j][r] + bv);
      } else if (OUT_MODE == 1) {
        int h = nf >> 6, hd = nf & (HD - 1);
        int mf = gRow + i * 16;
        int bb = mf / SS;
        int s0 = (mf % SS) + 8 * g;
        Pack8 pk;
#pragma unroll
        for (int r = 0; r < 8; ++r) pk.h[r] = (bf16)(acc[i][j][r] + bv);
        bf16* dst = (bf16*)Out + (((size_t)bb * HH + h) * HD + hd) * SS + s0;
        *(v4u*)dst = pk.u;
      } else {
        float* dst = (float*)Out + (size_t)(gRow + i * 16 + 8 * g) * Ndim + nf;
#pragma unroll
        for (int r = 0; r < 8; ++r)
          dst[(size_t)r * Ndim] = acc[i][j][r] + bv;
      }
    }
  }
}

// ---- flash attention, transposed compute: St = K Q^T, ctx^T = V^T P^T ----
// Block = 8 waves, all on one (b,h). K/V^T chunks staged into triple-buffered LDS
// via CDNA5 async-to-LDS copies (ASYNCcnt), one barrier per chunk. Each wave owns
// 32 query rows.
__global__ __launch_bounds__(256) void attn_kernel(const bf16* __restrict__ q,
    const bf16* __restrict__ k, const bf16* __restrict__ vt, bf16* __restrict__ ctx) {
  __shared__ __align__(16) bf16 kst[3][32][64];          // K chunk: 32 kv x 64 hd
  __shared__ __align__(16) bf16 vst[3][64][32];          // V^T chunk: 64 hd x 32 kv
  __shared__ __align__(16) unsigned int pbuf[8][2][16 * 20];  // per wave/qfrag P tile

  int tid = threadIdx.x;
  int lane = tid & 31, w = tid >> 5;
  int n15 = lane & 15, g = (lane >> 4) & 1;
  int bid = blockIdx.x;
  int qblk = bid & 7;          // 8 q-blocks per (b,h): 8 waves x 32 q = 256 q rows each
  int bh = bid >> 3;
  int h = bh & (HH - 1);
  int b = bh >> 4;
  int q0w = qblk * 256 + w * 32;

  const bf16* qh  = q  + (size_t)b * SS * DD + h * HD;
  const bf16* kh  = k  + (size_t)b * SS * DD + h * HD;
  const bf16* vth = vt + ((size_t)b * HH + h) * HD * SS;

  // cooperative staging coordinates (one async b128 each per thread per tile)
  int krow = tid >> 3, kseg = tid & 7;   // 32 rows x 8 x 16B
  int vrow = tid >> 2, vseg = tid & 3;   // 64 rows x 4 x 16B

  auto stage_async = [&](int ch, int buf) {
    int kv0 = ch * 32;
    async_load_b128(lds_addr(&kst[buf][krow][kseg * 8]),
                    kh + (size_t)(kv0 + krow) * DD + kseg * 8);
    async_load_b128(lds_addr(&vst[buf][vrow][vseg * 8]),
                    vth + (size_t)vrow * SS + kv0 + vseg * 8);
  };

  // Q fragments (B operand): 2 q-subtiles x 2 hd-chunks, held in registers
  v16bf bqf[2][2];
#pragma unroll
  for (int qf = 0; qf < 2; ++qf)
#pragma unroll
    for (int c = 0; c < 2; ++c)
      bqf[qf][c] = load_b_bf16(qh, DD, q0w + qf * 16, c * 32, lane);

  const float SC = 0.125f * 1.44269504f;  // (1/sqrt(HD)) * log2(e)
  float m_run[2] = {-1e30f, -1e30f};
  float l_run[2] = {0.f, 0.f};
  v8f zero = {};
  v8f cacc[2][4];
#pragma unroll
  for (int qf = 0; qf < 2; ++qf)
#pragma unroll
    for (int d = 0; d < 4; ++d) cacc[qf][d] = zero;

  stage_async(0, 0);  // prologue: chunk 0 in flight

  const int NCH = SS / 32;  // 64 key chunks
  for (int i = 0; i < NCH; ++i) {
    int buf = i % 3;
    if (i + 1 < NCH) {
      // chunk i+1 targets the buffer consumed at chunk i-2 (safe past last barrier);
      // its async latency overlaps this chunk's WMMA work
      stage_async(i + 1, (i + 1) % 3);
      asm volatile("s_wait_asynccnt 0x2" ::: "memory");  // chunk i resident in LDS
    } else {
      asm volatile("s_wait_asynccnt 0x0" ::: "memory");
    }
    __syncthreads();

#pragma unroll
    for (int qf = 0; qf < 2; ++qf) {
      // scores^T fragments (M = 16 kv, N = 16 q), K fragments from LDS
      v8f st[2];
#pragma unroll
      for (int sub = 0; sub < 2; ++sub) {
        v16bf aK0 = load_a_bf16(&kst[buf][0][0], 64, sub * 16, 0, lane);
        v16bf aK1 = load_a_bf16(&kst[buf][0][0], 64, sub * 16, 32, lane);
        st[sub] = wmma_bf16(aK0, bqf[qf][0], zero);
        st[sub] = wmma_bf16(aK1, bqf[qf][1], st[sub]);
      }
      // online softmax over kv; per-lane = one query column, halves synced via shfl_xor 16
      float smax = -1e30f;
#pragma unroll
      for (int sub = 0; sub < 2; ++sub)
#pragma unroll
        for (int r = 0; r < 8; ++r) smax = fmaxf(smax, st[sub][r]);
      smax = fmaxf(smax, __shfl_xor(smax, 16, 32));
      float m_new = fmaxf(m_run[qf], smax);
      float corr = __builtin_amdgcn_exp2f((m_run[qf] - m_new) * SC);
      float psum = 0.f;
#pragma unroll
      for (int sub = 0; sub < 2; ++sub)
#pragma unroll
        for (int r = 0; r < 8; ++r) {
          float p = __builtin_amdgcn_exp2f((st[sub][r] - m_new) * SC);
          st[sub][r] = p;
          psum += p;
        }
      psum += __shfl_xor(psum, 16, 32);
      l_run[qf] = l_run[qf] * corr + psum;
      m_run[qf] = m_new;
#pragma unroll
      for (int d = 0; d < 4; ++d)
#pragma unroll
        for (int r = 0; r < 8; ++r) cacc[qf][d][r] *= corr;
      // relayout P^T (C-layout) -> LDS as P row-major [q][kv], packed dwords
      unsigned int* prow = &pbuf[w][qf][0];
#pragma unroll
      for (int sub = 0; sub < 2; ++sub)
#pragma unroll
        for (int rp = 0; rp < 4; ++rp) {
          Pack2 pk;
          pk.h[0] = (bf16)st[sub][2 * rp];
          pk.h[1] = (bf16)st[sub][2 * rp + 1];
          prow[n15 * 20 + sub * 8 + g * 4 + rp] = pk.u;  // kv = sub*16 + 8g + 2rp
        }
    }

    // B-operand fragments of P (32 kv x 16 q) for both q-subtiles
    ABFrag bp[2];
#pragma unroll
    for (int qf = 0; qf < 2; ++qf) {
      unsigned int* prow = &pbuf[w][qf][0];
      bp[qf].u[0] = *(const v4u*)&prow[n15 * 20 + g * 8];
      bp[qf].u[1] = *(const v4u*)&prow[n15 * 20 + g * 8 + 4];
    }
    // ctx^T += Vt[16 hd x 32 kv] @ P[32 kv x 16 q]; V fragment reused for both qf
#pragma unroll
    for (int d = 0; d < 4; ++d) {
      v16bf av = load_a_bf16(&vst[buf][0][0], 32, d * 16, 0, lane);
#pragma unroll
      for (int qf = 0; qf < 2; ++qf)
        cacc[qf][d] = wmma_bf16(av, bp[qf].v, cacc[qf][d]);
    }
  }

#pragma unroll
  for (int qf = 0; qf < 2; ++qf) {
    float inv = 1.f / l_run[qf];
    bf16* cb = ctx + (size_t)(b * SS + q0w + qf * 16 + n15) * DD + h * HD;
#pragma unroll
    for (int d = 0; d < 4; ++d) {
      Pack8 pk;
#pragma unroll
      for (int r = 0; r < 8; ++r) pk.h[r] = (bf16)(cacc[qf][d][r] * inv);
      *(v4u*)(cb + d * 16 + g * 8) = pk.u;  // hd = d*16 + 8g + r (contiguous)
    }
  }
}

extern "C" void kernel_launch(void* const* d_in, const int* in_sizes, int n_in,
                              void* d_out, int out_size, void* d_ws, size_t ws_size,
                              hipStream_t stream) {
  (void)in_sizes; (void)n_in; (void)out_size; (void)ws_size;
  const float* query = (const float*)d_in[0];
  const float* key_  = (const float*)d_in[1];
  const float* value = (const float*)d_in[2];
  const float* Wq = (const float*)d_in[3];
  const float* bq = (const float*)d_in[4];
  const float* Wk = (const float*)d_in[5];
  const float* bk = (const float*)d_in[6];
  const float* Wv = (const float*)d_in[7];
  const float* bv = (const float*)d_in[8];
  const float* Wo = (const float*)d_in[9];
  const float* bo = (const float*)d_in[10];
  float* out = (float*)d_out;

  char* ws = (char*)d_ws;
  const size_t wbytes = (size_t)DD * DD * sizeof(bf16);  // 2 MB each
  const size_t xbytes = (size_t)MM * DD * sizeof(bf16);  // 8 MB each
  bf16* wq_t = (bf16*)ws; ws += wbytes;
  bf16* wk_t = (bf16*)ws; ws += wbytes;
  bf16* wv_t = (bf16*)ws; ws += wbytes;
  bf16* wo_t = (bf16*)ws; ws += wbytes;
  bf16* qb   = (bf16*)ws; ws += xbytes;   // [B,S,D] bf16
  bf16* kb   = (bf16*)ws; ws += xbytes;   // [B,S,D] bf16
  bf16* vtb  = (bf16*)ws; ws += xbytes;   // [B,H,HD,S] bf16
  bf16* ctxb = (bf16*)ws; ws += xbytes;   // [B,S,D] bf16
  // total workspace: 40 MB

  dim3 tgrid(DD / 32, DD / 32);
  wconv_kernel<<<tgrid, 256, 0, stream>>>(Wq, wq_t);
  wconv_kernel<<<tgrid, 256, 0, stream>>>(Wk, wk_t);
  wconv_kernel<<<tgrid, 256, 0, stream>>>(Wv, wv_t);
  wconv_kernel<<<tgrid, 256, 0, stream>>>(Wo, wo_t);

  dim3 ggrid(MM / 128, DD / 128);  // 32 x 8
  gemm_kernel<0, 0><<<ggrid, 256, 0, stream>>>(query, wq_t, bq, qb,  MM, DD, DD);
  gemm_kernel<0, 0><<<ggrid, 256, 0, stream>>>(key_,  wk_t, bk, kb,  MM, DD, DD);
  gemm_kernel<0, 1><<<ggrid, 256, 0, stream>>>(value, wv_t, bv, vtb, MM, DD, DD);

  attn_kernel<<<dim3(BB * HH * 8), 256, 0, stream>>>(qb, kb, vtb, ctxb);

  gemm_kernel<1, 2><<<ggrid, 256, 0, stream>>>(ctxb, wo_t, bo, out, MM, DD, DD);
}
